// SelfAttention_85736137162885
// MI455X (gfx1250) — matrix operand order repaired and checked by reading, not compiled, not run on hardware
//
#include <hip/hip_runtime.h>
#include <hip/hip_bf16.h>

// ---------------------------------------------------------------------------
// Self-attention for MI455X (gfx1250, wave32, WMMA).
//   x[2,2048,2048] f32, w_qkv[6144,2048] f32, w_o[2048,2048] f32 -> out f32.
// Strategy: bf16 WMMA (f32 accumulate) everywhere; working set (~134MB) is
// L2-resident (192MB), so GEMMs run from L2 with 64x32 per-wave tiles.
// ---------------------------------------------------------------------------

#define D_MODEL 2048
#define NHEAD   16
#define HDIM    128
#define SEQ     2048
#define BATCH   2
#define QKV_N   (3 * D_MODEL)   // 6144

typedef __attribute__((ext_vector_type(16))) __bf16       v16bf;
typedef __attribute__((ext_vector_type(8)))  float        v8f;
typedef __attribute__((ext_vector_type(4)))  unsigned int u32x4;

union FragBits { u32x4 q[2]; v16bf b; };

__device__ __forceinline__ unsigned short f32_to_bf16_rne(float f) {
  unsigned int u = __float_as_uint(f);
  u += 0x7FFFu + ((u >> 16) & 1u);     // round-to-nearest-even
  return (unsigned short)(u >> 16);
}

// A-fragment (16x32 bf16, MxK) of a row-major matrix, per ISA 7.12.2:
//   lane L: row = L&15 ; k chunks at (L>>4)*8 and (L>>4)*8+16 (8 bf16 each).
__device__ __forceinline__ v16bf load_frag_a(const unsigned short* base, int ld) {
  int lane = threadIdx.x & 31;
  int row  = lane & 15;
  int c0   = (lane >> 4) << 3;
  const unsigned short* p = base + (size_t)row * (size_t)ld + c0;
  FragBits f;
  f.q[0] = *(const u32x4*)(p);
  f.q[1] = *(const u32x4*)(p + 16);
  return f.b;
}

// B-fragment (32x16 bf16, KxN) where column n of B is row n of the given
// row-major matrix (i.e. B = Mrows^T), per ISA dense-B layout:
//   lane L: col = L&15 ; k chunks at (L>>4)*8 and (L>>4)*8+16 (8 bf16 each).
__device__ __forceinline__ v16bf load_frag_b(const unsigned short* base, int ld) {
  int lane = threadIdx.x & 31;
  int row  = lane & 15;          // output column n -> row of source matrix
  int c0   = (lane >> 4) << 3;   // k offset
  const unsigned short* p = base + (size_t)row * (size_t)ld + c0;
  FragBits f;
  f.q[0] = *(const u32x4*)(p);
  f.q[1] = *(const u32x4*)(p + 16);
  return f.b;
}

__device__ __forceinline__ v8f wmma_bf16(v16bf a, v16bf b, v8f c) {
  return __builtin_amdgcn_wmma_f32_16x16x32_bf16(false, a, false, b,
                                                 (short)0, c, false, false);
}

// ---------------------------------------------------------------------------
// Kernel 1: fp32 -> bf16 (4 elements / thread, b128 load + b64 store)
// ---------------------------------------------------------------------------
__global__ void __launch_bounds__(256)
convert_f32_bf16(const float* __restrict__ src, unsigned short* __restrict__ dst,
                 int n4) {
  int i = blockIdx.x * blockDim.x + threadIdx.x;
  if (i >= n4) return;
  float4 v = ((const float4*)src)[i];
  unsigned long long pk =
      (unsigned long long)f32_to_bf16_rne(v.x) |
      ((unsigned long long)f32_to_bf16_rne(v.y) << 16) |
      ((unsigned long long)f32_to_bf16_rne(v.z) << 32) |
      ((unsigned long long)f32_to_bf16_rne(v.w) << 48);
  ((unsigned long long*)dst)[i] = pk;
}

// ---------------------------------------------------------------------------
// Kernel 2/5: C[M,N] = A[M,K] @ B[N,K]^T   (bf16 in, f32 acc)
// One wave computes a 64x32 tile (4x2 WMMA subtiles): 8 WMMAs / 12 b128 loads
// per k-step, ~112 base VGPRs so the compiler can double-buffer fragments.
// ---------------------------------------------------------------------------
__global__ void __launch_bounds__(256)
gemm_xwt(const unsigned short* __restrict__ A,
         const unsigned short* __restrict__ B,
         void* __restrict__ C,
         int M, int N, int K, int ldc, int storeBF16) {
  int wave = (int)((blockIdx.x * blockDim.x + threadIdx.x) >> 5);
  int lane = threadIdx.x & 31;
  int ntn  = N >> 5;                 // 32-wide N tiles
  int tm   = wave / ntn;             // 64-tall M tile
  int tn   = wave % ntn;
  if (tm * 64 >= M) return;

  const unsigned short* Abase = A + (size_t)(tm * 64) * (size_t)K;
  const unsigned short* Bbase = B + (size_t)(tn * 32) * (size_t)K;

  v8f zero8 = {0.f, 0.f, 0.f, 0.f, 0.f, 0.f, 0.f, 0.f};
  v8f acc[4][2];
#pragma unroll
  for (int mi = 0; mi < 4; mi++) {
    acc[mi][0] = zero8;
    acc[mi][1] = zero8;
  }

  for (int k = 0; k < K; k += 32) {
    // Stream upcoming tiles into GL2 (global_prefetch_b8).
    __builtin_prefetch(Abase + k + 128, 0, 1);
    __builtin_prefetch(Bbase + k + 128, 0, 1);
    v16bf bf[2];
    bf[0] = load_frag_b(Bbase + k, K);
    bf[1] = load_frag_b(Bbase + (size_t)16 * K + k, K);
#pragma unroll
    for (int mi = 0; mi < 4; mi++) {
      v16bf a = load_frag_a(Abase + (size_t)(mi * 16) * K + k, K);
      acc[mi][0] = wmma_bf16(a, bf[0], acc[mi][0]);
      acc[mi][1] = wmma_bf16(a, bf[1], acc[mi][1]);
    }
  }

  int r0  = (lane >> 4) << 3;   // row group within subtile
  int col = lane & 15;
#pragma unroll
  for (int mi = 0; mi < 4; mi++) {
#pragma unroll
    for (int ni = 0; ni < 2; ni++) {
#pragma unroll
      for (int i = 0; i < 8; i++) {
        size_t r = (size_t)(tm * 64 + mi * 16 + r0 + i);
        size_t c = (size_t)(tn * 32 + ni * 16 + col);
        float v = acc[mi][ni][i];
        if (storeBF16)
          ((unsigned short*)C)[r * (size_t)ldc + c] = f32_to_bf16_rne(v);
        else
          ((float*)C)[r * (size_t)ldc + c] = v;
      }
    }
  }
}

// ---------------------------------------------------------------------------
// Kernel 3: repack V slice of qkv (bf16) into v_t[B,H,HD,T] so the P@V
// B-fragment is a contiguous row read.
// ---------------------------------------------------------------------------
__global__ void __launch_bounds__(256)
repack_v(const unsigned short* __restrict__ qkv, unsigned short* __restrict__ vt) {
  int idx  = blockIdx.x * blockDim.x + threadIdx.x;   // over B*H*HD*SEQ
  int t    = idx & (SEQ - 1);
  int rest = idx >> 11;
  int d    = rest & (HDIM - 1);
  int bh   = rest >> 7;
  int h    = bh & (NHEAD - 1);
  int b    = bh >> 4;
  vt[idx] = qkv[(size_t)(b * SEQ + t) * QKV_N + 2 * D_MODEL + h * HDIM + d];
}

// ---------------------------------------------------------------------------
// Kernel 4: per-wave flash attention. Each wave owns 16 query rows of one
// (b,h); online softmax; P re-swizzled C-layout -> A-layout via 1KB LDS.
// ---------------------------------------------------------------------------
__global__ void __launch_bounds__(256)
flash_attn(const unsigned short* __restrict__ qkv,
           const unsigned short* __restrict__ vt,
           unsigned short* __restrict__ y) {
  __shared__ unsigned short plds[8][16 * 32];   // per-wave P staging (bf16)

  int waveId = threadIdx.x >> 5;
  int lane   = threadIdx.x & 31;
  int gw     = blockIdx.x * 8 + waveId;
  int qt     = gw & (SEQ / 16 - 1);   // query tile within (b,h)
  int bh     = gw >> 7;
  int h      = bh & (NHEAD - 1);
  int b      = bh >> 4;
  const float scale = 0.08838834764831845f;   // 1/sqrt(128)

  const unsigned short* qbase = qkv + (size_t)(b * SEQ + qt * 16) * QKV_N + h * HDIM;
  const unsigned short* kbase = qkv + (size_t)(b * SEQ) * QKV_N + D_MODEL + h * HDIM;
  const unsigned short* vbase = vt + (size_t)(b * NHEAD + h) * HDIM * SEQ;

  // Q fragments: 16 rows x 128 = 4 k-chunks, loaded once.
  v16bf qf[4];
#pragma unroll
  for (int kc = 0; kc < 4; kc++) qf[kc] = load_frag_a(qbase + kc * 32, QKV_N);

  v8f zero8 = {0.f, 0.f, 0.f, 0.f, 0.f, 0.f, 0.f, 0.f};
  v8f o[8];
#pragma unroll
  for (int dt = 0; dt < 8; dt++) o[dt] = zero8;
  float m[8], l[8];
#pragma unroll
  for (int i = 0; i < 8; i++) { m[i] = -1e30f; l[i] = 0.f; }

  for (int j = 0; j < SEQ; j += 32) {   // 32 keys per step
    // ---- S = (Q K^T) * scale for this 16x32 block ----
    v8f s0 = zero8, s1 = zero8;
#pragma unroll
    for (int kc = 0; kc < 4; kc++) {
      v16bf kb0 = load_frag_b(kbase + (size_t)j * QKV_N + kc * 32, QKV_N);
      v16bf kb1 = load_frag_b(kbase + (size_t)(j + 16) * QKV_N + kc * 32, QKV_N);
      s0 = wmma_bf16(qf[kc], kb0, s0);
      s1 = wmma_bf16(qf[kc], kb1, s1);
    }

    // ---- online softmax (row stats shuffled across the 16-lane half) ----
#pragma unroll
    for (int i = 0; i < 8; i++) {
      float v0 = s0[i] * scale;
      float v1 = s1[i] * scale;
      float mx = fmaxf(v0, v1);
      mx = fmaxf(mx, __shfl_xor(mx, 1));
      mx = fmaxf(mx, __shfl_xor(mx, 2));
      mx = fmaxf(mx, __shfl_xor(mx, 4));
      mx = fmaxf(mx, __shfl_xor(mx, 8));
      float mnew  = fmaxf(m[i], mx);
      float alpha = __expf(m[i] - mnew);
      float p0 = __expf(v0 - mnew);
      float p1 = __expf(v1 - mnew);
      float rs = p0 + p1;
      rs += __shfl_xor(rs, 1);
      rs += __shfl_xor(rs, 2);
      rs += __shfl_xor(rs, 4);
      rs += __shfl_xor(rs, 8);
      l[i] = l[i] * alpha + rs;
      m[i] = mnew;
#pragma unroll
      for (int dt = 0; dt < 8; dt++) o[dt][i] *= alpha;
      // C-layout -> LDS [row][key] bf16
      int row = ((lane >> 4) << 3) + i;
      plds[waveId][row * 32 + (lane & 15)]      = f32_to_bf16_rne(p0);
      plds[waveId][row * 32 + 16 + (lane & 15)] = f32_to_bf16_rne(p1);
    }

    // ---- O += P @ V  (A from LDS in A-frag layout; B contiguous from v_t) --
    v16bf pa = load_frag_a(&plds[waveId][0], 32);
#pragma unroll
    for (int dt = 0; dt < 8; dt++) {
      v16bf bv = load_frag_b(vbase + (size_t)(dt * 16) * SEQ + j, SEQ);
      o[dt] = wmma_bf16(pa, bv, o[dt]);
    }
  }

  // ---- normalize + store y (bf16, row-major [B*T, D]) ----
#pragma unroll
  for (int i = 0; i < 8; i++) {
    float inv = 1.0f / l[i];
    int row = ((lane >> 4) << 3) + i;
#pragma unroll
    for (int dt = 0; dt < 8; dt++) {
      y[(size_t)(b * SEQ + qt * 16 + row) * D_MODEL + h * HDIM + dt * 16 + (lane & 15)] =
          f32_to_bf16_rne(o[dt][i] * inv);
    }
  }
}

// ---------------------------------------------------------------------------
// Launch. Workspace layout (bf16 elements), total exactly 128 MiB:
//   xb(8.4M) | wqkvb(12.6M) | wob(4.2M) | qkvb(25.2M) | vtb(8.4M) | yb(8.4M)
// ---------------------------------------------------------------------------
extern "C" void kernel_launch(void* const* d_in, const int* in_sizes, int n_in,
                              void* d_out, int out_size, void* d_ws, size_t ws_size,
                              hipStream_t stream) {
  const float* x     = (const float*)d_in[0];
  const float* w_qkv = (const float*)d_in[1];
  const float* w_o   = (const float*)d_in[2];
  float* out = (float*)d_out;

  unsigned short* xb    = (unsigned short*)d_ws;
  unsigned short* wqkvb = xb    + (size_t)BATCH * SEQ * D_MODEL;
  unsigned short* wob   = wqkvb + (size_t)QKV_N * D_MODEL;
  unsigned short* qkvb  = wob   + (size_t)D_MODEL * D_MODEL;
  unsigned short* vtb   = qkvb  + (size_t)BATCH * SEQ * QKV_N;
  unsigned short* yb    = vtb   + (size_t)BATCH * NHEAD * HDIM * SEQ;

  // 1) fp32 -> bf16 conversions
  {
    int n4 = BATCH * SEQ * D_MODEL / 4;
    convert_f32_bf16<<<(n4 + 255) / 256, 256, 0, stream>>>(x, xb, n4);
    n4 = QKV_N * D_MODEL / 4;
    convert_f32_bf16<<<(n4 + 255) / 256, 256, 0, stream>>>(w_qkv, wqkvb, n4);
    n4 = D_MODEL * D_MODEL / 4;
    convert_f32_bf16<<<(n4 + 255) / 256, 256, 0, stream>>>(w_o, wob, n4);
  }

  // 2) qkv = x @ w_qkv^T : M=4096, N=6144, K=2048 -> bf16
  {
    int waves = (BATCH * SEQ / 64) * (QKV_N / 32);   // 64 * 192 = 12288
    gemm_xwt<<<waves / 8, 256, 0, stream>>>(xb, wqkvb, qkvb,
                                            BATCH * SEQ, QKV_N, D_MODEL,
                                            QKV_N, /*storeBF16=*/1);
  }

  // 3) repack V transposed per head: [B,H,HD,T]
  {
    int n = BATCH * NHEAD * HDIM * SEQ;
    repack_v<<<n / 256, 256, 0, stream>>>(qkvb, vtb);
  }

  // 4) flash attention -> yb (bf16)
  {
    int waves = BATCH * NHEAD * (SEQ / 16);   // 4096
    flash_attn<<<waves / 8, 256, 0, stream>>>(qkvb, vtb, yb);
  }

  // 5) out = y @ w_o^T : M=4096, N=2048, K=2048 -> f32
  {
    int waves = (BATCH * SEQ / 64) * (D_MODEL / 32);  // 64 * 64 = 4096
    gemm_xwt<<<waves / 8, 256, 0, stream>>>(yb, wob, out,
                                            BATCH * SEQ, D_MODEL, D_MODEL,
                                            D_MODEL, /*storeBF16=*/0);
  }
}
